// LOCOSTLayerSelfAttention_90426241450005
// MI455X (gfx1250) — matrix operand order repaired and checked by reading, not compiled, not run on hardware
//
#include <hip/hip_runtime.h>
#include <hip/hip_bf16.h>
#include <math.h>

// Problem constants (match reference setup_inputs()).
constexpr int kB = 4;
constexpr int kL = 4096;
constexpr int kH = 1024;
constexpr int kN = 16;          // SSM state size
constexpr float kEPS = 1e-6f;

typedef __attribute__((ext_vector_type(16))) __bf16 v16bf;
typedef __attribute__((ext_vector_type(8)))  __bf16 v8bf;
typedef __attribute__((ext_vector_type(4)))  __bf16 v4bf;
typedef __attribute__((ext_vector_type(8)))  float  v8f;
typedef __attribute__((ext_vector_type(4)))  int    v4i;

// CDNA5 async global->LDS copy (ASYNCcnt path); guarded so the file compiles
// on toolchains that do not declare the builtin.  Probe-confirmed signature:
// (v4i addrspace(1)* gsrc, v4i addrspace(3)* ldsdst, imm offset, imm cpol).
#if defined(__has_builtin)
#  if __has_builtin(__builtin_amdgcn_global_load_async_to_lds_b128)
#    define USE_ASYNC_LDS 1
#  endif
#endif
#ifndef USE_ASYNC_LDS
#  define USE_ASYNC_LDS 0
#endif

#if USE_ASYNC_LDS
typedef __attribute__((address_space(1))) v4i* g1v4i_t;
typedef __attribute__((address_space(3))) v4i* l3v4i_t;
__device__ inline void async_copy16(const void* g, void* l) {
  __builtin_amdgcn_global_load_async_to_lds_b128(
      (g1v4i_t)const_cast<void*>(g), (l3v4i_t)l, 0, 0);
}
__device__ inline void async_wait0() {
#  if __has_builtin(__builtin_amdgcn_s_wait_asynccnt)
  __builtin_amdgcn_s_wait_asynccnt(0);
#  else
  asm volatile("s_wait_asynccnt 0x0" ::: "memory");
#  endif
}
#endif

// ---------------------------------------------------------------------------
// Kernel 1: RMSNorm row statistics.  rscale[b*L+l] = rsqrt(mean(x^2)+eps)*mask
// ---------------------------------------------------------------------------
__global__ __launch_bounds__(256)
void rms_rows_kernel(const float* __restrict__ hidden,
                     const float* __restrict__ mask,
                     float* __restrict__ rscale) {
  __shared__ float red[256];
  const int row = blockIdx.x;                       // b*L + l
  const float4 v = reinterpret_cast<const float4*>(hidden + (size_t)row * kH)[threadIdx.x];
  red[threadIdx.x] = v.x * v.x + v.y * v.y + v.z * v.z + v.w * v.w;
  __syncthreads();
  #pragma unroll
  for (int s = 128; s > 0; s >>= 1) {
    if (threadIdx.x < s) red[threadIdx.x] += red[threadIdx.x + s];
    __syncthreads();
  }
  if (threadIdx.x == 0)
    rscale[row] = rsqrtf(red[0] * (1.0f / kH) + kEPS) * mask[row];
}

// ---------------------------------------------------------------------------
// Kernel 2: normalize + transpose (B,L,H) -> (B,H,L), coalesced via LDS tile.
// ---------------------------------------------------------------------------
__global__ __launch_bounds__(256)
void norm_transpose_kernel(const float* __restrict__ hidden,
                           const float* __restrict__ rscale,
                           const float* __restrict__ nw,
                           float* __restrict__ u) {
  __shared__ float tile[32][33];
  const int b  = blockIdx.z;
  const int l0 = blockIdx.x * 32;
  const int h0 = blockIdx.y * 32;
  const int tx = threadIdx.x, ty = threadIdx.y;     // 32 x 8
  #pragma unroll
  for (int i = 0; i < 4; ++i) {
    const int l = l0 + ty + i * 8;
    tile[ty + i * 8][tx] = hidden[((size_t)b * kL + l) * kH + h0 + tx];
  }
  __syncthreads();
  #pragma unroll
  for (int i = 0; i < 4; ++i) {
    const int h = h0 + ty + i * 8;
    const int l = l0 + tx;
    u[((size_t)b * kH + h) * kL + l] =
        tile[tx][ty + i * 8] * rscale[(size_t)b * kL + l] * nw[h];
  }
}

// ---------------------------------------------------------------------------
// Kernel 3: S4D discretization (lam = exp(dt*A), lam^16, c2 = 2*C*(lam-1)/A)
// ---------------------------------------------------------------------------
__global__ __launch_bounds__(256)
void s4d_params_kernel(const float* __restrict__ log_dt,
                       const float* __restrict__ log_A_real,
                       const float* __restrict__ A_imag,
                       const float* __restrict__ C_re,
                       const float* __restrict__ C_im,
                       float* __restrict__ lam_re, float* __restrict__ lam_im,
                       float* __restrict__ lamT_re, float* __restrict__ lamT_im,
                       float* __restrict__ c2_re, float* __restrict__ c2_im) {
  const int i = blockIdx.x * blockDim.x + threadIdx.x;
  if (i >= kH * kN) return;
  const int h = i >> 4;
  const float dt = expf(log_dt[h]);
  const float ar = -expf(log_A_real[i]);
  const float ai = A_imag[i];
  const float dr = ar * dt, di = ai * dt;
  const float er = expf(dr);
  const float lr = er * cosf(di), li = er * sinf(di);
  const float nr = lr - 1.0f, ni = li;
  const float inv = 1.0f / (ar * ar + ai * ai);
  const float tr = (nr * ar + ni * ai) * inv;
  const float ti = (ni * ar - nr * ai) * inv;
  const float cr = C_re[i], ci = C_im[i];
  c2_re[i] = 2.0f * (cr * tr - ci * ti);
  c2_im[i] = 2.0f * (cr * ti + ci * tr);
  lam_re[i] = lr; lam_im[i] = li;
  float xr = lr, xi = li;                            // lam^16 by squaring
  #pragma unroll
  for (int s = 0; s < 4; ++s) {
    const float r2 = xr * xr - xi * xi;
    const float i2 = 2.0f * xr * xi;
    xr = r2; xi = i2;
  }
  lamT_re[i] = xr; lamT_im[i] = xi;
}

// ---------------------------------------------------------------------------
// Kernel 4: out_w fp32 -> bf16
// ---------------------------------------------------------------------------
__global__ __launch_bounds__(256)
void cvt_w_kernel(const float* __restrict__ w, __bf16* __restrict__ wbf, int n) {
  const int i = blockIdx.x * blockDim.x + threadIdx.x;
  if (i < n) wbf[i] = (__bf16)w[i];
}

// ---------------------------------------------------------------------------
// Kernel 5: diagonal-SSM causal conv as chunked parallel scan + GELU -> bf16.
// ---------------------------------------------------------------------------
__global__ __launch_bounds__(256)
void s4d_scan_kernel(const float* __restrict__ u,
                     const float* __restrict__ lam_re, const float* __restrict__ lam_im,
                     const float* __restrict__ lamT_re, const float* __restrict__ lamT_im,
                     const float* __restrict__ c2_re, const float* __restrict__ c2_im,
                     const float* __restrict__ Dp,
                     __bf16* __restrict__ ybf) {
  constexpr int T = 16;                              // timesteps per thread
  constexpr int NT = 256;                            // threads (chunks)
  __shared__ float2 sbuf[kN][NT];                    // 32 KB chunk states
  __shared__ float p_lr[kN], p_li[kN], p_tr[kN], p_ti[kN], p_cr[kN], p_ci[kN];

  const int bh = blockIdx.x;                         // b*H + h
  const int h  = bh & (kH - 1);
  const int t  = threadIdx.x;
  if (t < kN) {
    const int idx = h * kN + t;
    p_lr[t] = lam_re[idx];  p_li[t] = lam_im[idx];
    p_tr[t] = lamT_re[idx]; p_ti[t] = lamT_im[idx];
    p_cr[t] = c2_re[idx];   p_ci[t] = c2_im[idx];
  }
  __syncthreads();

  float uv[T];
  {
    const float4* up4 = reinterpret_cast<const float4*>(u + (size_t)bh * kL + t * T);
    #pragma unroll
    for (int q = 0; q < 4; ++q) {
      const float4 v = up4[q];
      uv[4 * q + 0] = v.x; uv[4 * q + 1] = v.y; uv[4 * q + 2] = v.z; uv[4 * q + 3] = v.w;
    }
  }

  // phase 1: chunk-local end state (zero initial condition)
  float sr[kN], si[kN];
  #pragma unroll
  for (int n = 0; n < kN; ++n) { sr[n] = 0.0f; si[n] = 0.0f; }
  #pragma unroll
  for (int i = 0; i < T; ++i) {
    const float uvi = uv[i];
    #pragma unroll
    for (int n = 0; n < kN; ++n) {
      const float r = p_lr[n] * sr[n] - p_li[n] * si[n] + uvi;
      si[n] = p_lr[n] * si[n] + p_li[n] * sr[n];
      sr[n] = r;
    }
  }
  #pragma unroll
  for (int n = 0; n < kN; ++n) sbuf[n][t] = make_float2(sr[n], si[n]);
  __syncthreads();

  // phase 2: exclusive prefix over 256 chunks (one lane per mode)
  if (t < kN) {
    float pr = 0.0f, pi = 0.0f;
    const float tr = p_tr[t], ti = p_ti[t];
    for (int c = 0; c < NT; ++c) {
      const float2 v = sbuf[t][c];
      sbuf[t][c] = make_float2(pr, pi);
      const float npr = tr * pr - ti * pi + v.x;
      pi = tr * pi + ti * pr + v.y;
      pr = npr;
    }
  }
  __syncthreads();

  // phase 3: replay with incoming state; readout + D*u + exact GELU
  float xr[kN], xi[kN];
  #pragma unroll
  for (int n = 0; n < kN; ++n) { const float2 v = sbuf[n][t]; xr[n] = v.x; xi[n] = v.y; }
  const float Dh = Dp[h];
  __bf16 yout[T];
  #pragma unroll
  for (int i = 0; i < T; ++i) {
    const float uvi = uv[i];
    float acc = 0.0f;
    #pragma unroll
    for (int n = 0; n < kN; ++n) {
      const float r = p_lr[n] * xr[n] - p_li[n] * xi[n] + uvi;
      xi[n] = p_lr[n] * xi[n] + p_li[n] * xr[n];
      xr[n] = r;
      acc += p_cr[n] * xr[n] - p_ci[n] * xi[n];
    }
    const float v = acc + Dh * uvi;
    yout[i] = (__bf16)(0.5f * v * (1.0f + erff(v * 0.70710678118654752f)));
  }
  // two 16B stores instead of 16 b16 stores
  v8bf* yp = reinterpret_cast<v8bf*>(ybf + (size_t)bh * kL + t * T);
  v8bf o0, o1;
  #pragma unroll
  for (int i = 0; i < 8; ++i) { o0[i] = yout[i]; o1[i] = yout[8 + i]; }
  yp[0] = o0; yp[1] = o1;
}

// ---------------------------------------------------------------------------
// Kernel 6: fused output projection + GLU + residual using WMMA bf16.
// Block: 256 threads (8 waves, 2(m) x 4(n)); tile 64(m)x64(n), BK=64
// -> 8 v_wmma_f32_16x16x32_bf16 per wave per K iteration (16 iterations).
// W tiles staged with async global->LDS b128 copies (ASYNCcnt); y tile staged
// TRANSPOSED via 4x4 register-block transpose (4x b64 loads + 4x b64 stores,
// all compile-time indices) so both A and B fragments are contiguous
// per-lane ds_load_b128 pairs.  LDS row stride padded to 72 elements (144 B).
// ---------------------------------------------------------------------------
__device__ inline v8f wmma_bf16(v16bf a, v16bf b, v8f c) {
  return __builtin_amdgcn_wmma_f32_16x16x32_bf16(false, a, false, b, (short)0, c, false, false);
}

constexpr int BM = 64, BN = 64, BK = 64;
constexpr int LDP = 72;                              // padded LDS row stride

// A fragment (ISA 7.12.2, 16-bit A 16x32): lane m = lane&15, half = lane>>4.
// Per lane: K = kk*32 + {half*8+0..7} and {16+half*8+0..7}  -> two 16B loads.
__device__ inline v16bf load_frag_a(const __bf16* tile, int mrow, int kk) {
  const int lane = threadIdx.x & 31;
  const int m    = mrow + (lane & 15);
  const int kb   = kk * 32 + ((lane >> 4) << 3);
  const __bf16* p = tile + (size_t)m * LDP + kb;
  const v8bf lo = *reinterpret_cast<const v8bf*>(p);
  const v8bf hi = *reinterpret_cast<const v8bf*>(p + 16);
  return __builtin_shufflevector(lo, hi, 0, 1, 2, 3, 4, 5, 6, 7,
                                 8, 9, 10, 11, 12, 13, 14, 15);
}

// B fragment from TRANSPOSED tile Bt[n][k]: lane n = lane&15, half = lane>>4.
// Per lane: K = kk*32 + {half*8+0..7} and {16+half*8+0..7} -> two 16B loads.
__device__ inline v16bf load_frag_b(const __bf16* tileT, int ncol, int kk) {
  const int lane = threadIdx.x & 31;
  const int n    = ncol + (lane & 15);
  const int kb   = kk * 32 + ((lane >> 4) << 3);
  const __bf16* p = tileT + (size_t)n * LDP + kb;
  const v8bf lo = *reinterpret_cast<const v8bf*>(p);
  const v8bf hi = *reinterpret_cast<const v8bf*>(p + 16);
  return __builtin_shufflevector(lo, hi, 0, 1, 2, 3, 4, 5, 6, 7,
                                 8, 9, 10, 11, 12, 13, 14, 15);
}

__global__ __launch_bounds__(256)
void out_proj_glu_kernel(const __bf16* __restrict__ ybf,
                         const __bf16* __restrict__ wbf,
                         const float* __restrict__ bias,
                         const float* __restrict__ hidden,
                         float* __restrict__ out) {
  __shared__ __bf16 As[BM * LDP];                    // a-rows of W   (m x k)
  __shared__ __bf16 Gs[BM * LDP];                    // gate-rows of W(m x k)
  __shared__ __bf16 Bt[BN * LDP];                    // y tile, TRANSPOSED (n x k)

  const int m0 = blockIdx.x * BM;                    // h tile
  const int l0 = blockIdx.y * BN;                    // l tile
  const int b  = blockIdx.z;
  const int tid  = threadIdx.x;
  const int lane = tid & 31;
  const int wave = tid >> 5;
  const int wm = wave & 1;                           // 0..1  (32 rows each)
  const int wn = wave >> 1;                          // 0..3  (16 cols each)

  const __bf16* wA = wbf + (size_t)m0 * kH;
  const __bf16* wG = wbf + (size_t)(m0 + kH) * kH;
  const __bf16* yB = ybf + (size_t)b * kH * kL;

  v8f acc_a[2] = {};
  v8f acc_g[2] = {};

  // W-tile copy geometry: 64 rows x 64 cols bf16, 256 threads,
  // 16 contiguous bf16 (32B) per thread, coalesced in global.
  const int cr = tid >> 2;                           // row 0..63
  const int cc = (tid & 3) * 16;                     // col 0,16,32,48
  // y-tile copy geometry: 4x4 block per thread (register transpose).
  const int gk = (tid >> 4) * 4;                     // k rows gk..gk+3
  const int c4 = (tid & 15) * 4;                     // col base

  for (int k0 = 0; k0 < kH; k0 += BK) {
    // --- W tiles (row-major in LDS), async global->LDS ------------------
    const __bf16* gA = &wA[(size_t)cr * kH + k0 + cc];
    const __bf16* gG = &wG[(size_t)cr * kH + k0 + cc];
    __bf16* sA = &As[cr * LDP + cc];
    __bf16* sG = &Gs[cr * LDP + cc];
#if USE_ASYNC_LDS
    async_copy16(gA, sA);           async_copy16(gA + 8, sA + 8);
    async_copy16(gG, sG);           async_copy16(gG + 8, sG + 8);
#else
    reinterpret_cast<v8bf*>(sA)[0] = reinterpret_cast<const v8bf*>(gA)[0];
    reinterpret_cast<v8bf*>(sA)[1] = reinterpret_cast<const v8bf*>(gA)[1];
    reinterpret_cast<v8bf*>(sG)[0] = reinterpret_cast<const v8bf*>(gG)[0];
    reinterpret_cast<v8bf*>(sG)[1] = reinterpret_cast<const v8bf*>(gG)[1];
#endif
    // --- y tile: 4x4 register-block transpose into Bt -------------------
    {
      v4bf rowv[4];
      #pragma unroll
      for (int rr = 0; rr < 4; ++rr)
        rowv[rr] = *reinterpret_cast<const v4bf*>(
            &yB[(size_t)(k0 + gk + rr) * kL + l0 + c4]);
      #pragma unroll
      for (int c2 = 0; c2 < 4; ++c2) {
        v4bf colv;
        #pragma unroll
        for (int rr = 0; rr < 4; ++rr) colv[rr] = rowv[rr][c2];
        *reinterpret_cast<v4bf*>(&Bt[(size_t)(c4 + c2) * LDP + gk]) = colv;
      }
      if (k0 + BK < kH)                              // warm caches for next tile
        __builtin_prefetch(&yB[(size_t)(k0 + BK + gk) * kL + l0 + c4], 0, 1);
    }
#if USE_ASYNC_LDS
    async_wait0();
#endif
    __syncthreads();

    // --- 8 WMMAs per wave ------------------------------------------------
    #pragma unroll
    for (int kk = 0; kk < 2; ++kk) {
      const v16bf bf = load_frag_b(Bt, wn * 16, kk);
      #pragma unroll
      for (int r = 0; r < 2; ++r) {
        const v16bf af = load_frag_a(As, wm * 32 + r * 16, kk);
        acc_a[r] = wmma_bf16(af, bf, acc_a[r]);
        const v16bf gf = load_frag_a(Gs, wm * 32 + r * 16, kk);
        acc_g[r] = wmma_bf16(gf, bf, acc_g[r]);
      }
    }
    __syncthreads();
  }

  // Epilogue: bias + GLU + residual, transposed write to (B,L,H).
  const int half = lane >> 4;
  const int n    = lane & 15;
  const int l    = l0 + wn * 16 + n;
  #pragma unroll
  for (int r = 0; r < 2; ++r) {
    const int hbase = m0 + wm * 32 + r * 16 + half * 8;
    const size_t o  = ((size_t)b * kL + l) * kH + hbase;
    #pragma unroll
    for (int j = 0; j < 8; ++j) {
      const float za  = acc_a[r][j] + bias[hbase + j];
      const float zg  = acc_g[r][j] + bias[hbase + j + kH];
      const float val = za / (1.0f + expf(-zg));     // a * sigmoid(g)
      out[o + j] = hidden[o + j] + val;
    }
  }
}

// ---------------------------------------------------------------------------
// Host-side launcher
// ---------------------------------------------------------------------------
extern "C" void kernel_launch(void* const* d_in, const int* in_sizes, int n_in,
                              void* d_out, int out_size, void* d_ws, size_t ws_size,
                              hipStream_t stream) {
  (void)in_sizes; (void)n_in; (void)out_size; (void)ws_size;

  const float* hidden  = (const float*)d_in[0];
  const float* mask    = (const float*)d_in[1];
  const float* norm_w  = (const float*)d_in[2];
  const float* log_dt  = (const float*)d_in[3];
  const float* log_Ar  = (const float*)d_in[4];
  const float* A_im    = (const float*)d_in[5];
  const float* C_re    = (const float*)d_in[6];
  const float* C_im    = (const float*)d_in[7];
  const float* Dp      = (const float*)d_in[8];
  const float* out_w   = (const float*)d_in[9];
  const float* out_b   = (const float*)d_in[10];
  float* out = (float*)d_out;

  const size_t BLH = (size_t)kB * kL * kH;
  const size_t HN  = (size_t)kH * kN;

  char* wsb = (char*)d_ws;
  size_t off = 0;
  auto take = [&](size_t bytes) -> char* {
    char* p = wsb + off;
    off = (off + bytes + 255) & ~(size_t)255;
    return p;
  };
  float*  u       = (float*)take(BLH * sizeof(float));       // (B,H,L) normalized input
  float*  rscale  = (float*)take((size_t)kB * kL * sizeof(float));
  float*  lam_re  = (float*)take(HN * sizeof(float));
  float*  lam_im  = (float*)take(HN * sizeof(float));
  float*  lamT_re = (float*)take(HN * sizeof(float));
  float*  lamT_im = (float*)take(HN * sizeof(float));
  float*  c2_re   = (float*)take(HN * sizeof(float));
  float*  c2_im   = (float*)take(HN * sizeof(float));
  __bf16* ybf     = (__bf16*)take(BLH * sizeof(__bf16));     // post-GELU activations
  __bf16* wbf     = (__bf16*)take((size_t)2 * kH * kH * sizeof(__bf16));

  rms_rows_kernel<<<kB * kL, 256, 0, stream>>>(hidden, mask, rscale);
  norm_transpose_kernel<<<dim3(kL / 32, kH / 32, kB), dim3(32, 8), 0, stream>>>(
      hidden, rscale, norm_w, u);
  s4d_params_kernel<<<(int)((HN + 255) / 256), 256, 0, stream>>>(
      log_dt, log_Ar, A_im, C_re, C_im,
      lam_re, lam_im, lamT_re, lamT_im, c2_re, c2_im);
  cvt_w_kernel<<<(int)((2 * kH * kH + 255) / 256), 256, 0, stream>>>(
      out_w, wbf, 2 * kH * kH);
  s4d_scan_kernel<<<kB * kH, 256, 0, stream>>>(
      u, lam_re, lam_im, lamT_re, lamT_im, c2_re, c2_im, Dp, ybf);
  out_proj_glu_kernel<<<dim3(kH / 64, kL / 64, kB), 256, 0, stream>>>(
      ybf, wbf, out_b, hidden, out);
}
